// PyGMOFuseMoE_4440996184165
// MI455X (gfx1250) — matrix operand order repaired and checked by reading, not compiled, not run on hardware
//
#include <hip/hip_runtime.h>

// ---------------------------------------------------------------------------
// Optional CDNA5 async global->LDS staging (ASYNCcnt path), guarded so the
// file compiles identically (register-staged fallback) if the builtin is
// absent from this toolchain.
// ---------------------------------------------------------------------------
#ifndef USE_ASYNC_LDS
#  if defined(__has_builtin)
#    if __has_builtin(__builtin_amdgcn_global_load_async_to_lds_b128)
#      define USE_ASYNC_LDS 1
#    else
#      define USE_ASYNC_LDS 0
#    endif
#  else
#    define USE_ASYNC_LDS 0
#  endif
#endif

// ---------------------------------------------------------------------------
// Types for CDNA5 WMMA (wave32): bf16 A/B fragments, f32 accumulator.
// ---------------------------------------------------------------------------
typedef __bf16          v16bf __attribute__((ext_vector_type(16)));
typedef unsigned short  v8u16 __attribute__((ext_vector_type(8)));
typedef float           v8f   __attribute__((ext_vector_type(8)));
typedef int             v4i   __attribute__((ext_vector_type(4)));

union Frag16 {
    v16bf v;
    v8u16 u[2];
};

// fp32 -> bf16 round-to-nearest-even (bit trick; inputs are well-formed finite)
__device__ __forceinline__ unsigned short f2bf(float f) {
    union { float f; unsigned int u; } x;
    x.f = f;
    unsigned int u = x.u;
    u += 0x7FFFu + ((u >> 16) & 1u);
    return (unsigned short)(u >> 16);
}

#if USE_ASYNC_LDS
// Builtin signature: (v4i addrspace(1)* src, v4i addrspace(3)* dst, Ii off, Ii cpol)
typedef __attribute__((address_space(1))) v4i* gas_ptr;  // global int4*
typedef __attribute__((address_space(3))) v4i* las_ptr;  // LDS int4*

// Global flat addresses are identity-mapped to AS1.
__device__ __forceinline__ gas_ptr to_gas(const void* p) {
    return (gas_ptr)(unsigned long long)(size_t)p;
}
// LDS flat addresses carry the aperture in bits[63:32]; addr[31:0] IS the
// LDS address (ISA 10.2), so truncation gives the AS3 pointer.
__device__ __forceinline__ las_ptr to_las(void* p) {
    return (las_ptr)(unsigned int)(size_t)p;
}
// Copy 64B (32 bf16) global -> LDS with 4 async b128 transfers.
__device__ __forceinline__ void stage_async64B(const unsigned short* g, unsigned short* l) {
    __builtin_amdgcn_global_load_async_to_lds_b128(to_gas(g),      to_las(l),      0, 0);
    __builtin_amdgcn_global_load_async_to_lds_b128(to_gas(g +  8), to_las(l +  8), 0, 0);
    __builtin_amdgcn_global_load_async_to_lds_b128(to_gas(g + 16), to_las(l + 16), 0, 0);
    __builtin_amdgcn_global_load_async_to_lds_b128(to_gas(g + 24), to_las(l + 24), 0, 0);
}
__device__ __forceinline__ void wait_async0() {
#if defined(__has_builtin)
#  if __has_builtin(__builtin_amdgcn_s_wait_asynccnt)
    __builtin_amdgcn_s_wait_asynccnt(0);
#  else
    asm volatile("s_wait_asynccnt 0x0" ::: "memory");
#  endif
#else
    asm volatile("s_wait_asynccnt 0x0" ::: "memory");
#endif
}
#endif // USE_ASYNC_LDS

// ---------------------------------------------------------------------------
// Elementwise fp32 -> bf16 convert (for x). count must be multiple of 1024.
// ---------------------------------------------------------------------------
__global__ __launch_bounds__(256) void convert_bf16_kernel(
    const float* __restrict__ src, unsigned short* __restrict__ dst)
{
    const size_t i = ((size_t)blockIdx.x * 256 + threadIdx.x) * 4;
    const float4 v = *(const float4*)(src + i);
    unsigned short o0 = f2bf(v.x), o1 = f2bf(v.y), o2 = f2bf(v.z), o3 = f2bf(v.w);
    unsigned long long pk = (unsigned long long)o0 | ((unsigned long long)o1 << 16) |
                            ((unsigned long long)o2 << 32) | ((unsigned long long)o3 << 48);
    *(unsigned long long*)(dst + i) = pk;
}

// ---------------------------------------------------------------------------
// Transpose + convert: Wt[n*K + k] = bf16(W[k*N + n]).  K,N multiples of 32.
// ---------------------------------------------------------------------------
__global__ __launch_bounds__(256) void transpose_convert_kernel(
    const float* __restrict__ W, unsigned short* __restrict__ Wt, int K, int N)
{
    __shared__ unsigned short tile[32][33];
    const int bk = blockIdx.y * 32;
    const int bn = blockIdx.x * 32;
    const int tx = threadIdx.x & 31;
    const int ty = threadIdx.x >> 5;   // 0..7
    #pragma unroll
    for (int i = 0; i < 4; i++) {
        const int k = ty + i * 8;
        tile[k][tx] = f2bf(W[(size_t)(bk + k) * N + bn + tx]);
    }
    __syncthreads();
    #pragma unroll
    for (int i = 0; i < 4; i++) {
        const int n = ty + i * 8;
        Wt[(size_t)(bn + n) * K + bk + tx] = tile[tx][n];
    }
}

// ---------------------------------------------------------------------------
// Tiled GEMM: C[M,N] = epilogue(A[M,K] * Bt[N,K]^T + bias[N])
//   A: bf16 row-major.  Bt: bf16 pre-transposed (k contiguous).
//   Block tile 128x128, BK=64 (2 WMMA k-steps), double-buffered LDS.
//   256 threads = 8 waves (2x4); each wave owns 64x32 = 4x2 WMMA 16x16 tiles.
// MODE 0: out fp32 = relu(acc + bias)            (gating hidden)
// MODE 1: out bf16 = relu(acc + bias)            (expert hidden h)
// MODE 2: out fp32 (+)= gates[m] * (acc + bias)  (gated combine)
// ---------------------------------------------------------------------------
template <int MODE, bool ACCUM>
__global__ __launch_bounds__(256) void gemm_wmma_bf16(
    const unsigned short* __restrict__ A,
    const unsigned short* __restrict__ Bt,
    const float* __restrict__ bias,
    const float* __restrict__ gates,   // stride-8 per row, pre-offset by expert
    void* __restrict__ Cptr,
    int N, int K)
{
    (void)gates;
    constexpr int BM = 128, BN = 128, BK = 64;
    constexpr int LDSS = BK + 8;       // 72 elements (144B): 16B-aligned, bank-spread

    __shared__ __align__(16) unsigned short As[2][BM * LDSS];
    __shared__ __align__(16) unsigned short Bs[2][BN * LDSS];

    const int t    = threadIdx.x;
    const int lane = t & 31;
    const int wid  = t >> 5;
    const int wm   = (wid >> 2) * 64;
    const int wn   = (wid & 3) * 32;
    const int h16  = lane >> 4;
    const int l15  = lane & 15;

    const int bm = blockIdx.y * BM;
    const int bn = blockIdx.x * BN;

    // staging map: 2 threads per 128-row, 32 contiguous bf16 (64B) each
    const int srow = t >> 1;
    const int scol = (t & 1) * 32;

    const unsigned short* Abase = A  + (size_t)(bm + srow) * K + scol;
    const unsigned short* Bbase = Bt + (size_t)(bn + srow) * K + scol;
    const int sLds = srow * LDSS + scol;

    v8f acc[4][2] = {};
    const int nk = K / BK;

#if USE_ASYNC_LDS
    // prologue: async-stage tile 0
    stage_async64B(Abase, &As[0][sLds]);
    stage_async64B(Bbase, &Bs[0][sLds]);
    wait_async0();
    __syncthreads();
#else
    v8u16 ra[4], rb[4];
    {
        const v8u16* Ag = (const v8u16*)Abase;
        const v8u16* Bg = (const v8u16*)Bbase;
        ra[0] = Ag[0]; ra[1] = Ag[1]; ra[2] = Ag[2]; ra[3] = Ag[3];
        rb[0] = Bg[0]; rb[1] = Bg[1]; rb[2] = Bg[2]; rb[3] = Bg[3];
        v8u16* pa = (v8u16*)&As[0][sLds];
        v8u16* pb = (v8u16*)&Bs[0][sLds];
        pa[0] = ra[0]; pa[1] = ra[1]; pa[2] = ra[2]; pa[3] = ra[3];
        pb[0] = rb[0]; pb[1] = rb[1]; pb[2] = rb[2]; pb[3] = rb[3];
    }
    __syncthreads();
#endif

    for (int kt = 0; kt < nk; kt++) {
        const int  cur  = kt & 1;
        const bool more = (kt + 1) < nk;
#if USE_ASYNC_LDS
        if (more) {
            stage_async64B(Abase + (size_t)(kt + 1) * BK, &As[1 - cur][sLds]);
            stage_async64B(Bbase + (size_t)(kt + 1) * BK, &Bs[1 - cur][sLds]);
        }
#else
        if (more) {
            const v8u16* Ag = (const v8u16*)(Abase + (size_t)(kt + 1) * BK);
            const v8u16* Bg = (const v8u16*)(Bbase + (size_t)(kt + 1) * BK);
            ra[0] = Ag[0]; ra[1] = Ag[1]; ra[2] = Ag[2]; ra[3] = Ag[3];
            rb[0] = Bg[0]; rb[1] = Bg[1]; rb[2] = Bg[2]; rb[3] = Bg[3];
        }
#endif
        // two WMMA k-steps from the current LDS buffers
        #pragma unroll
        for (int s = 0; s < 2; s++) {
            // A (16x32, 16-bit): lane half h, row m: K in [8h,8h+7] U [16+8h,16+8h+7]
            Frag16 a[4];
            #pragma unroll
            for (int i = 0; i < 4; i++) {
                const unsigned short* p =
                    &As[cur][(wm + i * 16 + l15) * LDSS + s * 32 + h16 * 8];
                a[i].u[0] = *(const v8u16*)p;
                a[i].u[1] = *(const v8u16*)(p + 16);
            }
            // B (32x16, 16-bit): lane half h, col n: K in [8h,8h+7] U [16+8h,16+8h+7]
            Frag16 b[2];
            #pragma unroll
            for (int j = 0; j < 2; j++) {
                const unsigned short* p =
                    &Bs[cur][(wn + j * 16 + l15) * LDSS + s * 32 + h16 * 8];
                b[j].u[0] = *(const v8u16*)p;
                b[j].u[1] = *(const v8u16*)(p + 16);
            }
            #pragma unroll
            for (int i = 0; i < 4; i++)
                #pragma unroll
                for (int j = 0; j < 2; j++)
                    acc[i][j] = __builtin_amdgcn_wmma_f32_16x16x32_bf16(
                        false, a[i].v, false, b[j].v, (short)0, acc[i][j], false, false);
        }
#if USE_ASYNC_LDS
        if (more) wait_async0();
#else
        if (more) {
            v8u16* pa = (v8u16*)&As[1 - cur][sLds];
            v8u16* pb = (v8u16*)&Bs[1 - cur][sLds];
            pa[0] = ra[0]; pa[1] = ra[1]; pa[2] = ra[2]; pa[3] = ra[3];
            pb[0] = rb[0]; pb[1] = rb[1]; pb[2] = rb[2]; pb[3] = rb[3];
        }
#endif
        __syncthreads();
    }

    // ---- epilogue: C/D layout lane(l15)=N, VGPR r -> M = 8*h16 + r ----
    #pragma unroll
    for (int i = 0; i < 4; i++) {
        #pragma unroll
        for (int j = 0; j < 2; j++) {
            const int   gn  = bn + wn + j * 16 + l15;
            const int   gm0 = bm + wm + i * 16 + 8 * h16;
            const float bv  = bias[gn];
            #pragma unroll
            for (int r = 0; r < 8; r++) {
                const int m = gm0 + r;
                float v = acc[i][j][r] + bv;
                if constexpr (MODE == 0) {
                    ((float*)Cptr)[(size_t)m * N + gn] = fmaxf(v, 0.0f);
                } else if constexpr (MODE == 1) {
                    ((unsigned short*)Cptr)[(size_t)m * N + gn] = f2bf(fmaxf(v, 0.0f));
                } else {
                    const float g = gates[(size_t)m * 8];
                    v *= g;
                    float* o = (float*)Cptr + (size_t)m * N + gn;
                    if constexpr (ACCUM) v += *o;
                    *o = v;
                }
            }
        }
    }
}

// ---------------------------------------------------------------------------
// Gating head: logits = gh @ gW2 + gb2 (K=2048, E=8), softmax over E.
// One wave per row; writes gates to both output slots.
// ---------------------------------------------------------------------------
__global__ __launch_bounds__(256) void gating_softmax_kernel(
    const float* __restrict__ gh, const float* __restrict__ W2,
    const float* __restrict__ b2, float* __restrict__ g1,
    float* __restrict__ g2, int K)
{
    const int row  = blockIdx.x * 8 + (threadIdx.x >> 5);
    const int lane = threadIdx.x & 31;
    const float* xr = gh + (size_t)row * K;

    float acc[8] = {0.f, 0.f, 0.f, 0.f, 0.f, 0.f, 0.f, 0.f};
    for (int k = lane; k < K; k += 32) {
        const float  xv = xr[k];
        const float4 w0 = *(const float4*)(W2 + (size_t)k * 8);
        const float4 w1 = *(const float4*)(W2 + (size_t)k * 8 + 4);
        acc[0] += xv * w0.x; acc[1] += xv * w0.y; acc[2] += xv * w0.z; acc[3] += xv * w0.w;
        acc[4] += xv * w1.x; acc[5] += xv * w1.y; acc[6] += xv * w1.z; acc[7] += xv * w1.w;
    }
    #pragma unroll
    for (int e = 0; e < 8; e++) {
        #pragma unroll
        for (int off = 16; off > 0; off >>= 1)
            acc[e] += __shfl_xor(acc[e], off, 32);
        acc[e] += b2[e];
    }
    float mx = acc[0];
    #pragma unroll
    for (int e = 1; e < 8; e++) mx = fmaxf(mx, acc[e]);
    float s = 0.f;
    #pragma unroll
    for (int e = 0; e < 8; e++) { acc[e] = __expf(acc[e] - mx); s += acc[e]; }
    const float inv = 1.0f / s;
    #pragma unroll
    for (int e = 0; e < 8; e++) acc[e] *= inv;

    if (lane == 0) {
        float4 lo, hi;
        lo.x = acc[0]; lo.y = acc[1]; lo.z = acc[2]; lo.w = acc[3];
        hi.x = acc[4]; hi.y = acc[5]; hi.z = acc[6]; hi.w = acc[7];
        float4* p1 = (float4*)(g1 + (size_t)row * 8);
        p1[0] = lo; p1[1] = hi;
        float4* p2 = (float4*)(g2 + (size_t)row * 8);
        p2[0] = lo; p2[1] = hi;
    }
}

// ---------------------------------------------------------------------------
// Launcher
// d_in order: x, gW1, gb1, gW2, gb2, eW1, eb1, eW2, eb2 (all fp32)
// d_out: combined[8192*1024] | gates[8192*8] | gates[8192*8]
// d_ws: xb bf16 16MB | gh fp32 64MB | h bf16 64MB | wt bf16 8MB  => 152MB
// ---------------------------------------------------------------------------
extern "C" void kernel_launch(void* const* d_in, const int* in_sizes, int n_in,
                              void* d_out, int out_size, void* d_ws, size_t ws_size,
                              hipStream_t stream)
{
    (void)in_sizes; (void)n_in; (void)out_size; (void)ws_size;

    const int B = 8192, DIN = 1024, DH = 4096, DO = 1024, E = 8, GH = 2048;

    const float* x   = (const float*)d_in[0];
    const float* gW1 = (const float*)d_in[1];
    const float* gb1 = (const float*)d_in[2];
    const float* gW2 = (const float*)d_in[3];
    const float* gb2 = (const float*)d_in[4];
    const float* eW1 = (const float*)d_in[5];
    const float* eb1 = (const float*)d_in[6];
    const float* eW2 = (const float*)d_in[7];
    const float* eb2 = (const float*)d_in[8];

    float* out      = (float*)d_out;
    float* combined = out;                    // [B, DO]
    float* gates1   = out + (size_t)B * DO;   // [B, E]
    float* gates2   = gates1 + (size_t)B * E; // [B, E]

    char* ws = (char*)d_ws;
    unsigned short* xb = (unsigned short*)ws;                  // B*DIN bf16  (16MB)
    ws += (size_t)B * DIN * 2;
    float* gh = (float*)ws;                                    // B*GH fp32   (64MB)
    ws += (size_t)B * GH * 4;
    unsigned short* hbuf = (unsigned short*)ws;                // B*DH bf16   (64MB)
    ws += (size_t)B * DH * 2;
    unsigned short* wt = (unsigned short*)ws;                  // up to DH*DO bf16 (8MB)

    const dim3 blk(256);

    // 0) x -> bf16
    convert_bf16_kernel<<<(B * DIN) / (256 * 4), blk, 0, stream>>>(x, xb);

    // 1) gating hidden: gh = relu(x @ gW1 + gb1)
    transpose_convert_kernel<<<dim3(GH / 32, DIN / 32), blk, 0, stream>>>(gW1, wt, DIN, GH);
    gemm_wmma_bf16<0, false>
        <<<dim3(GH / 128, B / 128), blk, 0, stream>>>(xb, wt, gb1, nullptr, gh, GH, DIN);

    // 2) gates = softmax(gh @ gW2 + gb2), written to both output slots
    gating_softmax_kernel<<<B / 8, blk, 0, stream>>>(gh, gW2, gb2, gates1, gates2, GH);

    // 3) experts (sequential on stream): h = relu(x @ eW1[e] + eb1[e]) -> bf16 ws,
    //    combined (+)= gates[:,e] * (h @ eW2[e] + eb2[e])
    for (int e = 0; e < E; e++) {
        transpose_convert_kernel<<<dim3(DH / 32, DIN / 32), blk, 0, stream>>>(
            eW1 + (size_t)e * DIN * DH, wt, DIN, DH);
        gemm_wmma_bf16<1, false>
            <<<dim3(DH / 128, B / 128), blk, 0, stream>>>(
                xb, wt, eb1 + (size_t)e * DH, nullptr, hbuf, DH, DIN);

        transpose_convert_kernel<<<dim3(DO / 32, DH / 32), blk, 0, stream>>>(
            eW2 + (size_t)e * DH * DO, wt, DH, DO);
        if (e == 0) {
            gemm_wmma_bf16<2, false>
                <<<dim3(DO / 128, B / 128), blk, 0, stream>>>(
                    hbuf, wt, eb2 + (size_t)e * DO, gates1 + e, combined, DO, DH);
        } else {
            gemm_wmma_bf16<2, true>
                <<<dim3(DO / 128, B / 128), blk, 0, stream>>>(
                    hbuf, wt, eb2 + (size_t)e * DO, gates1 + e, combined, DO, DH);
        }
    }
}
